// LlamaDecoderLayer_37933151158660
// MI455X (gfx1250) — compile-verified, matrix operand order checked by reading, not run on hardware
//
#include <hip/hip_runtime.h>
#include <hip/hip_bf16.h>

// ---------------------------------------------------------------------------
// Llama decoder layer for MI455X (gfx1250, wave32, WMMA).
// All GEMMs + attention matmuls use v_wmma_f32_16x16x32_bf16 (f32 accum).
// Weights transposed+converted to bf16 once; GEMM staging is pure b128
// load/store, double-buffered LDS (1 barrier / K-step), software-pipelined.
// ---------------------------------------------------------------------------

#define TT   2048
#define HH   4096
#define NQH  32
#define NKVH 8
#define HD   128
#define FF   14336
#define GQ   (NQH / NKVH)     // 4
#define QD   (NQH * HD)       // 4096
#define KVD  (NKVH * HD)      // 1024

typedef __bf16        v16bf __attribute__((ext_vector_type(16)));
typedef float         v8f   __attribute__((ext_vector_type(8)));
typedef unsigned int  v4u   __attribute__((ext_vector_type(4)));

union Frag { v16bf v; v4u q[2]; };

__device__ __forceinline__ unsigned short f2bf(float f) {
  unsigned int u = __float_as_uint(f);
  u += 0x7FFFu + ((u >> 16) & 1u);   // round-to-nearest-even
  return (unsigned short)(u >> 16);
}

__device__ __forceinline__ unsigned int pack2bf(float lo, float hi) {
  return (unsigned int)f2bf(lo) | ((unsigned int)f2bf(hi) << 16);
}

__device__ __forceinline__ v8f zero8() {
  v8f z;
#pragma unroll
  for (int i = 0; i < 8; ++i) z[i] = 0.0f;
  return z;
}

__device__ __forceinline__ v8f wmma_bf16(Frag a, Frag b, v8f c) {
  // 8 args: (neg_a, A, neg_b, B, c_mod, C, reuse_a, reuse_b)
  return __builtin_amdgcn_wmma_f32_16x16x32_bf16(false, a.v, false, b.v,
                                                 (short)0, c, false, false);
}

// ---------------------------------------------------------------------------
// One-time weight prep: in (K x N f32, row-major) -> out (N x K bf16).
// Tiled 32x32 LDS transpose, coalesced on both sides.
// ---------------------------------------------------------------------------
__global__ __launch_bounds__(256) void transpose_f32_to_bf16(
    const float* __restrict__ in, unsigned short* __restrict__ out,
    int K, int N) {
  __shared__ unsigned short tile[32][33];
  const int n0 = blockIdx.x * 32, k0 = blockIdx.y * 32;
  const int tx = threadIdx.x & 31, ty = threadIdx.x >> 5;  // ty: 0..7
#pragma unroll
  for (int i = 0; i < 4; ++i) {
    const int k = ty + i * 8;
    tile[k][tx] = f2bf(in[(size_t)(k0 + k) * N + (n0 + tx)]);
  }
  __syncthreads();
#pragma unroll
  for (int i = 0; i < 4; ++i) {
    const int n = ty + i * 8;
    out[(size_t)(n0 + n) * K + (k0 + tx)] = tile[tx][n];
  }
}

// ---------------------------------------------------------------------------
// GEMM: C[MxN] = A[MxK](bf16,row-major) @ Bt[NxK](bf16,row-major = B^T).
// Block tile 128x128, BK=32, 8 waves. Wave wv: m-tiles {2*(wv&3), +1},
// n-tiles (wv>>2)*4 .. +3  -> 8 WMMAs/wave/K-step, B frags reused 2x.
// Double-buffered LDS: one barrier per K-step; all fragments loaded into
// distinct registers before the WMMA chain (staggered dscnt waits).
// Epilogue modes: 0: f32 C   1: bf16 C   2: bf16( acc * silu(aux) )
//                 3: bf16 transposed C (C[n][m]) with packed b128 stores.
// ---------------------------------------------------------------------------
#define BM 128
#define BN 128
#define BK 32

__global__ __launch_bounds__(256) void gemm_bf16_wmma(
    const unsigned short* __restrict__ Ab, const unsigned short* __restrict__ Bt,
    float* __restrict__ Cf, unsigned short* __restrict__ Cb,
    const float* __restrict__ aux, int M, int N, int K, int mode) {
  __shared__ __attribute__((aligned(16))) unsigned short As[2][BM * BK]; // [m][k]
  __shared__ __attribute__((aligned(16))) unsigned short Bs[2][BN * BK]; // [n][k]

  const int bm0  = blockIdx.y * BM;
  const int bn0  = blockIdx.x * BN;
  const int tid  = threadIdx.x;
  const int lane = tid & 31;
  const int wv   = tid >> 5;
  const int wmp  = wv & 3;     // m-pair
  const int nh   = wv >> 2;    // n-half
  const int hf   = lane >> 4;  // K-half selector (ISA A/B frag layout)
  const int l16  = lane & 15;

  v8f acc[2][4];
#pragma unroll
  for (int mi = 0; mi < 2; ++mi)
#pragma unroll
    for (int t = 0; t < 4; ++t) acc[mi][t] = zero8();

  // Staging: 4096 bf16 per tile = 512 x b128; thread handles chunks tid, tid+256.
  const int r0  = tid >> 2;            // 0..63
  const int r1  = r0 + 64;             // 64..127
  const int c8  = (tid & 3) * 8;       // 0,8,16,24
  const unsigned short* agp0 = Ab + (size_t)(bm0 + r0) * K + c8;
  const unsigned short* agp1 = Ab + (size_t)(bm0 + r1) * K + c8;
  const unsigned short* bgp0 = Bt + (size_t)(bn0 + r0) * K + c8;
  const unsigned short* bgp1 = Bt + (size_t)(bn0 + r1) * K + c8;
  const int soff0 = r0 * BK + c8;
  const int soff1 = r1 * BK + c8;

  const int a0off = (wmp * 32 + l16) * BK + hf * 8;
  const int a1off = a0off + 16 * BK;

  // Tile 0: load + store into buffer 0 (no barrier needed before first sync).
  v4u pa0 = *(const v4u*)agp0, pa1 = *(const v4u*)agp1;
  v4u pb0 = *(const v4u*)bgp0, pb1 = *(const v4u*)bgp1;
  *(v4u*)&As[0][soff0] = pa0;
  *(v4u*)&As[0][soff1] = pa1;
  *(v4u*)&Bs[0][soff0] = pb0;
  *(v4u*)&Bs[0][soff1] = pb1;

  const int nIter = K / BK;
  for (int it = 0; it < nIter; ++it) {
    const int cur = it & 1;
    const bool more = (it + 1) < nIter;
    if (more) {  // prefetch next tile; overlaps barrier + compute
      const int ko = (it + 1) * BK;
      pa0 = *(const v4u*)(agp0 + ko);
      pa1 = *(const v4u*)(agp1 + ko);
      pb0 = *(const v4u*)(bgp0 + ko);
      pb1 = *(const v4u*)(bgp1 + ko);
    }
    __syncthreads();  // buf[cur] stores visible; all waves done reading buf[1-cur]

    // Load ALL fragments into distinct registers, then the 8-WMMA chain.
    Frag af0, af1, bfr[4];
    af0.q[0] = *(const v4u*)&As[cur][a0off];
    af0.q[1] = *(const v4u*)&As[cur][a0off + 16];
    af1.q[0] = *(const v4u*)&As[cur][a1off];
    af1.q[1] = *(const v4u*)&As[cur][a1off + 16];
#pragma unroll
    for (int t = 0; t < 4; ++t) {
      const int boff = ((nh * 4 + t) * 16 + l16) * BK + hf * 8;
      bfr[t].q[0] = *(const v4u*)&Bs[cur][boff];
      bfr[t].q[1] = *(const v4u*)&Bs[cur][boff + 16];
    }
#pragma unroll
    for (int t = 0; t < 4; ++t) {
      acc[0][t] = wmma_bf16(af0, bfr[t], acc[0][t]);
      acc[1][t] = wmma_bf16(af1, bfr[t], acc[1][t]);
    }

    if (more) {  // stage next tile into the other buffer
      const int nxt = 1 - cur;
      *(v4u*)&As[nxt][soff0] = pa0;
      *(v4u*)&As[nxt][soff1] = pa1;
      *(v4u*)&Bs[nxt][soff0] = pb0;
      *(v4u*)&Bs[nxt][soff1] = pb1;
    }
  }

  // C/D layout: element v of lane -> row M = v + hf*8, col N = l16.
#pragma unroll
  for (int mi = 0; mi < 2; ++mi) {
    const int rowb = bm0 + (wmp * 2 + mi) * 16 + hf * 8;
#pragma unroll
    for (int t = 0; t < 4; ++t) {
      const int col = bn0 + (nh * 4 + t) * 16 + l16;
      if (mode == 0) {
#pragma unroll
        for (int v = 0; v < 8; ++v)
          Cf[(size_t)(rowb + v) * N + col] = acc[mi][t][v];
      } else if (mode == 1) {
#pragma unroll
        for (int v = 0; v < 8; ++v)
          Cb[(size_t)(rowb + v) * N + col] = f2bf(acc[mi][t][v]);
      } else if (mode == 2) {
#pragma unroll
        for (int v = 0; v < 8; ++v) {
          const size_t id = (size_t)(rowb + v) * N + col;
          const float g = aux[id];
          const float sig = 1.0f / (1.0f + __expf(-g));
          Cb[id] = f2bf(acc[mi][t][v] * g * sig);
        }
      } else {  // mode 3: C^T bf16, 8 consecutive rows -> one b128 store
        v4u pk;
#pragma unroll
        for (int w = 0; w < 4; ++w)
          pk[w] = pack2bf(acc[mi][t][2 * w], acc[mi][t][2 * w + 1]);
        *(v4u*)&Cb[(size_t)col * M + rowb] = pk;
      }
    }
  }
}

// ---------------------------------------------------------------------------
// res = a + b ; norm(bf16) = rmsnorm(res) * w.  One block (256 thr) per row.
// ---------------------------------------------------------------------------
__global__ __launch_bounds__(256) void add_rmsnorm_kernel(
    const float* __restrict__ a, const float* __restrict__ b,
    const float* __restrict__ w, float* __restrict__ res,
    unsigned short* __restrict__ norm) {
  const int row = blockIdx.x;
  const size_t base = (size_t)row * HH;
  const int tid = threadIdx.x;
  __shared__ float red[8];

  float vals[HH / 256];
  float ss = 0.f;
#pragma unroll
  for (int i = 0; i < HH / 256; ++i) {
    const int c = tid + i * 256;
    const float x = a[base + c] + b[base + c];
    vals[i] = x;
    res[base + c] = x;
    ss += x * x;
  }
#pragma unroll
  for (int o = 16; o >= 1; o >>= 1) ss += __shfl_xor(ss, o, 32);
  if ((tid & 31) == 0) red[tid >> 5] = ss;
  __syncthreads();
  float tot = 0.f;
#pragma unroll
  for (int i = 0; i < 8; ++i) tot += red[i];
  const float inv = rsqrtf(tot / (float)HH + 1e-6f);
#pragma unroll
  for (int i = 0; i < HH / 256; ++i) {
    const int c = tid + i * 256;
    norm[base + c] = f2bf(vals[i] * inv * w[c]);
  }
}

// ---------------------------------------------------------------------------
// RoPE on (T, nheads, HD) f32 -> bf16. Grid (T, nheads), 64 threads.
// ---------------------------------------------------------------------------
__global__ __launch_bounds__(64) void rope_to_bf16_kernel(
    const float* __restrict__ x, const int* __restrict__ pos,
    unsigned short* __restrict__ out, int nheads) {
  const int t = blockIdx.x;
  const int hh = blockIdx.y;
  const int j = threadIdx.x;  // pair index 0..63
  const size_t base = ((size_t)t * nheads + hh) * HD;
  const float inv_freq = __expf(-((float)(2 * j) / (float)HD) * 9.2103403719761836f);
  const float f = (float)pos[t] * inv_freq;
  float s, c;
  __sincosf(f, &s, &c);
  const float x1 = x[base + j];
  const float x2 = x[base + j + HD / 2];
  out[base + j]          = f2bf(x1 * c - x2 * s);
  out[base + j + HD / 2] = f2bf(x2 * c + x1 * s);
}

// ---------------------------------------------------------------------------
// Causal GQA flash attention, all-WMMA.
// Grid (T/64, NQH), 128 threads (4 waves). Wave wv owns 16 query rows.
// vt is per-head transposed V: vt[kvh][d][t] (bf16) -> b128 staging.
// Output written bf16 (A-operand of o-proj).
// ---------------------------------------------------------------------------
__global__ __launch_bounds__(128) void attn_kernel(
    const unsigned short* __restrict__ qb, const unsigned short* __restrict__ kb,
    const unsigned short* __restrict__ vt, unsigned short* __restrict__ out) {
  const int qh   = blockIdx.y;
  const int kvh  = qh / GQ;
  const int wgi0 = blockIdx.x * 64;
  const int tid  = threadIdx.x;
  const int lane = tid & 31;
  const int wv   = tid >> 5;
  const int i0   = wgi0 + wv * 16;
  const int hf   = lane >> 4;
  const int l16  = lane & 15;
  const int rowm = hf * 8;

  __shared__ __attribute__((aligned(16))) unsigned short Vt[HD * 32];    // [d][k]
  __shared__ __attribute__((aligned(16))) unsigned short Pl[4][16 * 32]; // per-wave

  // Q fragments: this wave's 16 rows, full HD (4 k-chunks), kept in regs.
  Frag qf[4];
  const unsigned short* qrow = qb + ((size_t)(i0 + l16) * NQH + qh) * HD;
#pragma unroll
  for (int kk = 0; kk < 4; ++kk) {
    qf[kk].q[0] = *(const v4u*)&qrow[kk * 32 + hf * 8];
    qf[kk].q[1] = *(const v4u*)&qrow[kk * 32 + 16 + hf * 8];
  }

  v8f acc[8];
#pragma unroll
  for (int d = 0; d < 8; ++d) acc[d] = zero8();
  float mrow[8], lrow[8];
#pragma unroll
  for (int v = 0; v < 8; ++v) { mrow[v] = -3.0e38f; lrow[v] = 0.f; }

  const float sc = 0.08838834764831845f;  // 1/sqrt(128)
  const int jend = wgi0 + 64;

  // V^T staging addresses: 512 b128 chunks / 128 threads = 4 per thread.
  const unsigned short* vbase = vt + (size_t)kvh * HD * TT;

  for (int j0 = 0; j0 < jend; j0 += 32) {
    __syncthreads();  // previous chunk's Vt reads complete
#pragma unroll
    for (int it = 0; it < 4; ++it) {
      const int chunk = tid + it * 128;     // 0..511
      const int d  = chunk >> 2;            // 0..127
      const int k8 = (chunk & 3) * 8;       // 0,8,16,24
      *(v4u*)&Vt[d * 32 + k8] = *(const v4u*)&vbase[(size_t)d * TT + j0 + k8];
    }
    __syncthreads();

    // S = Q @ K^T, two 16-col tiles; K rows are [n][k]-contiguous in global.
    v8f s0 = zero8(), s1 = zero8();
#pragma unroll
    for (int kk = 0; kk < 4; ++kk) {
      const unsigned short* k0r =
          kb + ((size_t)(j0 + l16) * NKVH + kvh) * HD + kk * 32;
      Frag kf0;
      kf0.q[0] = *(const v4u*)&k0r[hf * 8];
      kf0.q[1] = *(const v4u*)&k0r[16 + hf * 8];
      s0 = wmma_bf16(qf[kk], kf0, s0);
      const unsigned short* k1r =
          kb + ((size_t)(j0 + 16 + l16) * NKVH + kvh) * HD + kk * 32;
      Frag kf1;
      kf1.q[0] = *(const v4u*)&k1r[hf * 8];
      kf1.q[1] = *(const v4u*)&k1r[16 + hf * 8];
      s1 = wmma_bf16(qf[kk], kf1, s1);
    }

    // Scale + causal mask + online softmax (row lives in one 16-lane half).
    float alpha[8];
    unsigned short* pw = &Pl[wv][0];
#pragma unroll
    for (int v = 0; v < 8; ++v) {
      const int ig = i0 + rowm + v;
      float a0 = s0[v] * sc;
      float a1 = s1[v] * sc;
      a0 = ((j0 + l16) <= ig) ? a0 : -3.0e38f;
      a1 = ((j0 + 16 + l16) <= ig) ? a1 : -3.0e38f;
      float mx = fmaxf(a0, a1);
#pragma unroll
      for (int o = 8; o >= 1; o >>= 1) mx = fmaxf(mx, __shfl_xor(mx, o, 32));
      const float mnew = fmaxf(mrow[v], mx);
      alpha[v] = __expf(mrow[v] - mnew);
      mrow[v] = mnew;
      const float p0 = __expf(a0 - mnew);
      const float p1 = __expf(a1 - mnew);
      float sum = p0 + p1;
#pragma unroll
      for (int o = 8; o >= 1; o >>= 1) sum += __shfl_xor(sum, o, 32);
      lrow[v] = lrow[v] * alpha[v] + sum;
      pw[(rowm + v) * 32 + l16]      = f2bf(p0);   // C-layout -> P[m][k]
      pw[(rowm + v) * 32 + 16 + l16] = f2bf(p1);
    }
#pragma unroll
    for (int d = 0; d < 8; ++d)
#pragma unroll
      for (int v = 0; v < 8; ++v) acc[d][v] *= alpha[v];

    // O += P(16x32) @ V(32x128).  (LDS in-order within a wave.)
    Frag pf;
    pf.q[0] = *(const v4u*)&pw[l16 * 32 + hf * 8];
    pf.q[1] = *(const v4u*)&pw[l16 * 32 + 16 + hf * 8];
#pragma unroll
    for (int d = 0; d < 8; ++d) {
      const int voff = (d * 16 + l16) * 32 + hf * 8;
      Frag vf;
      vf.q[0] = *(const v4u*)&Vt[voff];
      vf.q[1] = *(const v4u*)&Vt[voff + 16];
      acc[d] = wmma_bf16(pf, vf, acc[d]);
    }
  }

  // O /= l ; store bf16 to (T, NQH*HD).
#pragma unroll
  for (int d = 0; d < 8; ++d) {
    const int col = qh * HD + d * 16 + l16;
#pragma unroll
    for (int v = 0; v < 8; ++v)
      out[(size_t)(i0 + rowm + v) * QD + col] = f2bf(acc[d][v] / lrow[v]);
  }
}

// ---------------------------------------------------------------------------
extern "C" void kernel_launch(void* const* d_in, const int* in_sizes, int n_in,
                              void* d_out, int out_size, void* d_ws, size_t ws_size,
                              hipStream_t stream) {
  const int*   positions = (const int*)d_in[0];
  const float* hidden    = (const float*)d_in[1];
  const float* residual  = (const float*)d_in[2];
  const float* w_q       = (const float*)d_in[3];
  const float* w_k       = (const float*)d_in[4];
  const float* w_v       = (const float*)d_in[5];
  const float* w_o       = (const float*)d_in[6];
  const float* w_gate    = (const float*)d_in[7];
  const float* w_up      = (const float*)d_in[8];
  const float* w_down    = (const float*)d_in[9];
  const float* ln1_w     = (const float*)d_in[10];
  const float* ln2_w     = (const float*)d_in[11];

  float* out_x    = (float*)d_out;
  float* out_res2 = out_x + (size_t)TT * HH;

  const size_t TH  = (size_t)TT * HH;    // 8,388,608
  const size_t TKV = (size_t)TT * KVD;   // 2,097,152
  const size_t TFF = (size_t)TT * FF;    // 29,360,128
  const size_t WQO = (size_t)HH * QD;    // 16,777,216
  const size_t WKV = (size_t)HH * KVD;   // 4,194,304
  const size_t WFB = (size_t)HH * FF;    // 58,720,256

  // f32 scratch
  float* ws    = (float*)d_ws;
  float* res1  = ws;            // hs + residual
  float* q32   = res1 + TH;     // f32 Q pre-RoPE; reused as o-proj output
  float* k32   = q32 + TH;      // f32 K pre-RoPE
  float* gatef = k32 + TKV;     // f32 gate activation
  // bf16 scratch
  unsigned short* x1b   = (unsigned short*)(gatef + TFF);
  unsigned short* x2b   = x1b + TH;
  unsigned short* attnb = x2b + TH;
  unsigned short* hb    = attnb + TH;   // up * silu(gate)
  unsigned short* qbb   = hb + TFF;
  unsigned short* kbb   = qbb + TH;
  unsigned short* vtb   = kbb + TKV;    // per-head transposed V: [kvh][d][t]
  unsigned short* wqT   = vtb + TKV;    // weights, transposed to N x K bf16
  unsigned short* wkT   = wqT + WQO;
  unsigned short* wvT   = wkT + WKV;
  unsigned short* woT   = wvT + WKV;
  unsigned short* wgT   = woT + WQO;
  unsigned short* wuT   = wgT + WFB;
  unsigned short* wdT   = wuT + WFB;

  // 0) One-pass weight transpose+convert: (K x N f32) -> (N x K bf16)
  transpose_f32_to_bf16<<<dim3(QD / 32, HH / 32), 256, 0, stream>>>(w_q, wqT, HH, QD);
  transpose_f32_to_bf16<<<dim3(KVD / 32, HH / 32), 256, 0, stream>>>(w_k, wkT, HH, KVD);
  transpose_f32_to_bf16<<<dim3(KVD / 32, HH / 32), 256, 0, stream>>>(w_v, wvT, HH, KVD);
  transpose_f32_to_bf16<<<dim3(HH / 32, QD / 32), 256, 0, stream>>>(w_o, woT, QD, HH);
  transpose_f32_to_bf16<<<dim3(FF / 32, HH / 32), 256, 0, stream>>>(w_gate, wgT, HH, FF);
  transpose_f32_to_bf16<<<dim3(FF / 32, HH / 32), 256, 0, stream>>>(w_up, wuT, HH, FF);
  transpose_f32_to_bf16<<<dim3(HH / 32, FF / 32), 256, 0, stream>>>(w_down, wdT, FF, HH);

  // 1) hs = hidden + residual ; x1 = rmsnorm(hs, ln1) (bf16)
  add_rmsnorm_kernel<<<TT, 256, 0, stream>>>(hidden, residual, ln1_w, res1, x1b);

  // 2) QKV projections (V straight to per-head transposed bf16)
  gemm_bf16_wmma<<<dim3(QD / BN, TT / BM), 256, 0, stream>>>(
      x1b, wqT, q32, nullptr, nullptr, TT, QD, HH, 0);
  gemm_bf16_wmma<<<dim3(KVD / BN, TT / BM), 256, 0, stream>>>(
      x1b, wkT, k32, nullptr, nullptr, TT, KVD, HH, 0);
  gemm_bf16_wmma<<<dim3(KVD / BN, TT / BM), 256, 0, stream>>>(
      x1b, wvT, nullptr, vtb, nullptr, TT, KVD, HH, 3);

  // 3) RoPE -> bf16
  rope_to_bf16_kernel<<<dim3(TT, NQH), 64, 0, stream>>>(q32, positions, qbb, NQH);
  rope_to_bf16_kernel<<<dim3(TT, NKVH), 64, 0, stream>>>(k32, positions, kbb, NKVH);

  // 4) Causal GQA flash attention (bf16 out)
  attn_kernel<<<dim3(TT / 64, NQH), 128, 0, stream>>>(qbb, kbb, vtb, attnb);

  // 5) O projection (f32, into q32 scratch)
  gemm_bf16_wmma<<<dim3(HH / BN, TT / BM), 256, 0, stream>>>(
      attnb, woT, q32, nullptr, nullptr, TT, HH, QD, 0);

  // 6) res2 = oproj + res1 ; x2 = rmsnorm(res2, ln2) (bf16)
  add_rmsnorm_kernel<<<TT, 256, 0, stream>>>(q32, res1, ln2_w, out_res2, x2b);

  // 7) MLP: gate (f32) ; up with fused silu-mul epilogue (bf16 h) ; down (f32)
  gemm_bf16_wmma<<<dim3(FF / BN, TT / BM), 256, 0, stream>>>(
      x2b, wgT, gatef, nullptr, nullptr, TT, FF, HH, 0);
  gemm_bf16_wmma<<<dim3(FF / BN, TT / BM), 256, 0, stream>>>(
      x2b, wuT, nullptr, hb, gatef, TT, FF, HH, 2);
  gemm_bf16_wmma<<<dim3(HH / BN, TT / BM), 256, 0, stream>>>(
      hb, wdT, out_x, nullptr, nullptr, TT, HH, FF, 0);
}
